// DeepGraphSAGE_4312147165750
// MI455X (gfx1250) — compile-verified, matrix operand order checked by reading, not run on hardware
//
#include <hip/hip_runtime.h>
#include <hip/hip_bf16.h>

#define NPG_C 400
#define F_IN_C 200
#define H_C 64

typedef __attribute__((ext_vector_type(16))) _Float16 v16h;
typedef __attribute__((ext_vector_type(8)))  float    v8f;

// ------------------------------------------------- per-graph degree (LDS)
__global__ __launch_bounds__(256) void deg_pergraph(const int* __restrict__ dstI,
                                                    float* __restrict__ deg, int epg) {
  __shared__ float dacc[NPG_C];
  const int g = blockIdx.x, t = threadIdx.x;
  for (int i = t; i < NPG_C; i += 256) dacc[i] = 0.0f;
  __syncthreads();
  const int ebase = g * epg, nodeBase = g * NPG_C;
  for (int e = t; e < epg; e += 256)
    atomicAdd(&dacc[dstI[ebase + e] - nodeBase], 1.0f);
  __syncthreads();
  for (int i = t; i < NPG_C; i += 256) deg[nodeBase + i] = dacc[i];
}

// ------------------------------------------------------ WMMA GEMM  C = A @ W
// C[n,64] = A[n,K] @ W[K,64], f16 multiply / f32 accumulate
// (v_wmma_f32_16x16x32_f16).  K is compile-time -> no exec-masked predication;
// tail chunk (K=200) uses clamped unconditional loads + selects.
// Fragment layouts per CDNA5 ISA 7.12.2.
template <int K>
__global__ __launch_bounds__(256) void gemm_wmma_f16(
    const float* __restrict__ A, const float* __restrict__ W,
    float* __restrict__ C, int n) {
  __shared__ float wlds[K * H_C];          // stage W once per block
  for (int i = threadIdx.x; i < K * H_C; i += 256) wlds[i] = W[i];
  __syncthreads();

  const int lane = threadIdx.x & 31;
  const int wave = threadIdx.x >> 5;
  const int hi   = lane >> 4;              // K-half selector (0/1)
  const int l16  = lane & 15;
  const int rowBase = (blockIdx.x * 8 + wave) * 16;
  if (rowBase >= n) return;                // uniform per wave (grid is exact here)
  const int row = rowBase + l16;
  const float* __restrict__ arow = A + (size_t)row * K;

  constexpr int KFULL = (K / 32) * 32;
  constexpr bool TAIL = (K % 32) != 0;

  v8f acc[4] = {};
#pragma unroll
  for (int k0 = 0; k0 < KFULL; k0 += 32) {
    // A fragment: elems 0..7 <- arow[k0+hi*8 .. +7], 8..15 <- arow[k0+16+hi*8 ..]
    const float* ap = arow + k0 + hi * 8;
    const float4 f0 = *(const float4*)(ap);
    const float4 f1 = *(const float4*)(ap + 4);
    const float4 f2 = *(const float4*)(ap + 16);
    const float4 f3 = *(const float4*)(ap + 20);
    v16h a;
    a[0] = (_Float16)f0.x;  a[1] = (_Float16)f0.y;  a[2] = (_Float16)f0.z;  a[3] = (_Float16)f0.w;
    a[4] = (_Float16)f1.x;  a[5] = (_Float16)f1.y;  a[6] = (_Float16)f1.z;  a[7] = (_Float16)f1.w;
    a[8] = (_Float16)f2.x;  a[9] = (_Float16)f2.y;  a[10] = (_Float16)f2.z; a[11] = (_Float16)f2.w;
    a[12] = (_Float16)f3.x; a[13] = (_Float16)f3.y; a[14] = (_Float16)f3.z; a[15] = (_Float16)f3.w;
#pragma unroll
    for (int c = 0; c < 4; ++c) {
      const int col = c * 16 + l16;
      v16h b;
#pragma unroll
      for (int e = 0; e < 8; ++e) {
        b[e]     = (_Float16)wlds[(k0 + hi * 8 + e) * H_C + col];
        b[e + 8] = (_Float16)wlds[(k0 + 16 + hi * 8 + e) * H_C + col];
      }
      acc[c] = __builtin_amdgcn_wmma_f32_16x16x32_f16(false, a, false, b,
                                                      (short)0, acc[c], false, false);
    }
  }
  if constexpr (TAIL) {
    // K=200: only K rows KFULL..K-1 are valid -> only hi==0 lanes, elems 0..7.
    const float* ap = arow + KFULL;        // clamped (ignores hi) -> always in-bounds
    const float4 f0 = *(const float4*)(ap);
    const float4 f1 = *(const float4*)(ap + 4);
    const float hz = (hi == 0) ? 1.0f : 0.0f;   // select, not branch
    v16h a = {};
    a[0] = (_Float16)(f0.x * hz); a[1] = (_Float16)(f0.y * hz);
    a[2] = (_Float16)(f0.z * hz); a[3] = (_Float16)(f0.w * hz);
    a[4] = (_Float16)(f1.x * hz); a[5] = (_Float16)(f1.y * hz);
    a[6] = (_Float16)(f1.z * hz); a[7] = (_Float16)(f1.w * hz);
#pragma unroll
    for (int c = 0; c < 4; ++c) {
      const int col = c * 16 + l16;
      v16h b = {};
#pragma unroll
      for (int e = 0; e < 8; ++e)
        b[e] = (_Float16)(wlds[(KFULL + e) * H_C + col] * hz);
      acc[c] = __builtin_amdgcn_wmma_f32_16x16x32_f16(false, a, false, b,
                                                      (short)0, acc[c], false, false);
    }
  }
#pragma unroll
  for (int c = 0; c < 4; ++c)
#pragma unroll
    for (int r = 0; r < 8; ++r)
      C[(size_t)(rowBase + r + hi * 8) * H_C + c * 16 + l16] = acc[c][r];
}

// ----------------- per-graph fused: zero + LDS scatter-add + SAGE combine
// one workgroup per graph; accumulator (400x64 f32 = 100KB) lives in LDS,
// edges for graph g are contiguous [g*epg, (g+1)*epg).
__global__ __launch_bounds__(1024, 1) void sage_agg_combine(
    const float* __restrict__ lin_l,      // projected src features (h @ Wl)
    const float* __restrict__ lin_r,      // h @ Wr
    const float* __restrict__ bias,
    const float* __restrict__ residual,   // nullptr for layer 1
    const float* __restrict__ deg,
    const int* __restrict__ srcI, const int* __restrict__ dstI,
    float* __restrict__ out, int epg) {
  __shared__ float acc[NPG_C * H_C];
  const int g = blockIdx.x, t = threadIdx.x;
  const int nodeBase = g * NPG_C;
  for (int i = t; i < NPG_C * H_C; i += 1024) acc[i] = 0.0f;
  __syncthreads();
  const int ebase = g * epg;
  const int part = t & 15;                 // 16 threads / edge, float4 each
  for (int e = t >> 4; e < epg; e += 64) {
    const int s = srcI[ebase + e];                       // global node id
    const int d = dstI[ebase + e] - nodeBase;            // local node id
    const float4 v = *(const float4*)(lin_l + (size_t)s * H_C + part * 4);
    float* dp = acc + d * H_C + part * 4;
    atomicAdd(dp + 0, v.x); atomicAdd(dp + 1, v.y);
    atomicAdd(dp + 2, v.z); atomicAdd(dp + 3, v.w);
  }
  __syncthreads();
  for (int i = t; i < NPG_C * H_C; i += 1024) {
    const int node = i >> 6, f = i & 63;
    const size_t gi = (size_t)(nodeBase + node) * H_C + f;
    const float invd = 1.0f / fmaxf(deg[nodeBase + node], 1.0f);
    float v = fmaxf(acc[i] * invd + bias[f] + lin_r[gi], 0.0f);
    if (residual) v += residual[gi];
    out[gi] = v;
  }
}

// ---------------- per-graph fused: sum-agg of x3 + GraphConv score (H -> 1)
__global__ __launch_bounds__(1024, 1) void score_pergraph(
    const float* __restrict__ x3,
    const int* __restrict__ srcI, const int* __restrict__ dstI,
    const float* __restrict__ Wpr, const float* __restrict__ bpr,
    const float* __restrict__ Wpo,
    float* __restrict__ score, int epg) {
  __shared__ float acc[NPG_C * H_C];
  const int g = blockIdx.x, t = threadIdx.x;
  const int nodeBase = g * NPG_C;
  for (int i = t; i < NPG_C * H_C; i += 1024) acc[i] = 0.0f;
  __syncthreads();
  const int ebase = g * epg;
  const int part = t & 15;
  for (int e = t >> 4; e < epg; e += 64) {
    const int s = srcI[ebase + e];
    const int d = dstI[ebase + e] - nodeBase;
    const float4 v = *(const float4*)(x3 + (size_t)s * H_C + part * 4);
    float* dp = acc + d * H_C + part * 4;
    atomicAdd(dp + 0, v.x); atomicAdd(dp + 1, v.y);
    atomicAdd(dp + 2, v.z); atomicAdd(dp + 3, v.w);
  }
  __syncthreads();
  if (t < NPG_C) {
    float s = bpr[0];
    const float* xr = x3 + (size_t)(nodeBase + t) * H_C;
    const int rot = t & 63;                // rotate to avoid LDS bank conflicts
    for (int i = 0; i < H_C; ++i) {
      const int f = (i + rot) & 63;
      s += acc[t * H_C + f] * Wpr[f] + xr[f] * Wpo[f];
    }
    score[nodeBase + t] = s;
  }
}

// ------------- per-graph: top-k set -> tanh-weighted mean pool -> log-softmax
__global__ __launch_bounds__(512) void pool_kernel(
    const float* __restrict__ score, const float* __restrict__ x3,
    const float* __restrict__ Wlin, const float* __restrict__ blin,
    float* __restrict__ out, int kSel) {
  __shared__ float s_sc[NPG_C];
  __shared__ float s_w[NPG_C];
  __shared__ float s_pool[H_C];
  __shared__ float s_logit[2];
  const int g = blockIdx.x, t = threadIdx.x;

  if (t < NPG_C) s_sc[t] = score[(size_t)g * NPG_C + t];
  __syncthreads();
  if (t < NPG_C) {
    const float my = s_sc[t];
    int rank = 0;
    for (int j = 0; j < NPG_C; ++j) {
      const float sj = s_sc[j];
      rank += (sj > my) || (sj == my && j < t);   // jax top_k tie-break
    }
    s_w[t] = (rank < kSel) ? tanhf(my) : 0.0f;
  }
  __syncthreads();
  if (t < H_C) {
    float a = 0.0f;
    const float* xg = x3 + (size_t)g * NPG_C * H_C;
    for (int j = 0; j < NPG_C; ++j) a += s_w[j] * xg[(size_t)j * H_C + t];
    s_pool[t] = a / (float)kSel;
  }
  __syncthreads();
  if (t < 2) {
    float a = blin[t];
    for (int f = 0; f < H_C; ++f) a += s_pool[f] * Wlin[f * 2 + t];
    s_logit[t] = a;
  }
  __syncthreads();
  if (t == 0) {
    const float l0 = s_logit[0], l1 = s_logit[1];
    const float m = fmaxf(l0, l1);
    const float lse = m + logf(expf(l0 - m) + expf(l1 - m));
    out[g * 2 + 0] = l0 - lse;
    out[g * 2 + 1] = l1 - lse;
  }
}

// ---------------------------------------------------------------- host driver
extern "C" void kernel_launch(void* const* d_in, const int* in_sizes, int n_in,
                              void* d_out, int out_size, void* d_ws, size_t ws_size,
                              hipStream_t stream) {
  (void)n_in; (void)out_size; (void)ws_size;
  const float* x    = (const float*)d_in[0];
  const int*   edge = (const int*)d_in[1];
  const float* W1l  = (const float*)d_in[3];
  const float* W1r  = (const float*)d_in[4];
  const float* b1   = (const float*)d_in[5];
  const float* W2l  = (const float*)d_in[6];
  const float* W2r  = (const float*)d_in[7];
  const float* b2   = (const float*)d_in[8];
  const float* W3l  = (const float*)d_in[9];
  const float* W3r  = (const float*)d_in[10];
  const float* b3   = (const float*)d_in[11];
  const float* Wpr  = (const float*)d_in[12];
  const float* bpr  = (const float*)d_in[13];
  const float* Wpo  = (const float*)d_in[14];
  const float* Wlin = (const float*)d_in[15];
  const float* blin = (const float*)d_in[16];
  float* out = (float*)d_out;

  const int N = in_sizes[0] / F_IN_C;
  const int E = in_sizes[1] / 2;
  const int B = N / NPG_C;
  const int EPG = E / B;
  const int kSel = NPG_C / 2;
  const int* srcI = edge;        // edge_index row 0
  const int* dstI = edge + E;    // edge_index row 1

  float* p = (float*)d_ws;
  float* deg    = p; p += N;
  float* scoreB = p; p += N;
  float* tmp_l  = p; p += (size_t)N * H_C;
  float* tmp_r  = p; p += (size_t)N * H_C;
  float* x1     = p; p += (size_t)N * H_C;
  float* x2     = p; p += (size_t)N * H_C;
  float* x3     = p; p += (size_t)N * H_C;

  const int gemmBlocks = (N + 127) / 128;

  deg_pergraph<<<B, 256, 0, stream>>>(dstI, deg, EPG);

  // layer 1: project-then-aggregate (mean_agg(x)@W == mean_agg(x@W))
  gemm_wmma_f16<F_IN_C><<<gemmBlocks, 256, 0, stream>>>(x, W1l, tmp_l, N);
  gemm_wmma_f16<F_IN_C><<<gemmBlocks, 256, 0, stream>>>(x, W1r, tmp_r, N);
  sage_agg_combine<<<B, 1024, 0, stream>>>(tmp_l, tmp_r, b1, nullptr, deg,
                                           srcI, dstI, x1, EPG);
  // layer 2 (+ residual x1)
  gemm_wmma_f16<H_C><<<gemmBlocks, 256, 0, stream>>>(x1, W2l, tmp_l, N);
  gemm_wmma_f16<H_C><<<gemmBlocks, 256, 0, stream>>>(x1, W2r, tmp_r, N);
  sage_agg_combine<<<B, 1024, 0, stream>>>(tmp_l, tmp_r, b2, x1, deg,
                                           srcI, dstI, x2, EPG);
  // layer 3 (+ residual x2)
  gemm_wmma_f16<H_C><<<gemmBlocks, 256, 0, stream>>>(x2, W3l, tmp_l, N);
  gemm_wmma_f16<H_C><<<gemmBlocks, 256, 0, stream>>>(x2, W3r, tmp_r, N);
  sage_agg_combine<<<B, 1024, 0, stream>>>(tmp_l, tmp_r, b3, x2, deg,
                                           srcI, dstI, x3, EPG);

  // SAGPool score (add-aggregation) fused per graph
  score_pergraph<<<B, 1024, 0, stream>>>(x3, srcI, dstI, Wpr, bpr, Wpo, scoreB, EPG);

  // top-k select, tanh-weighted mean pool, classify, log-softmax
  pool_kernel<<<B, 512, 0, stream>>>(scoreB, x3, Wlin, blin, out, kSel);
}
